// GINFeatures_84164179132425
// MI455X (gfx1250) — matrix operand order repaired and checked by reading, not compiled, not run on hardware
//
#include <hip/hip_runtime.h>
#include <hip/hip_bf16.h>

// ---------------- types ----------------
typedef __attribute__((ext_vector_type(16))) __bf16 v16bf;
typedef __attribute__((ext_vector_type(8)))  __bf16 v8bf;
typedef __attribute__((ext_vector_type(8)))  float  v8f;

#define N_NODES   50000
#define N_EDGES   800000
#define N_GRAPHS  64
#define D_IN      128
#define HID       256

__device__ __forceinline__ __bf16 f2bf(float f) { return (__bf16)f; }

// ---------------- simple data movement ----------------
__global__ void copy_f32_vec4(const float* __restrict__ src, float* __restrict__ dst, int n4) {
    int i = blockIdx.x * blockDim.x + threadIdx.x;
    if (i < n4) ((float4*)dst)[i] = ((const float4*)src)[i];
}

__global__ void cvt_f32_bf16(const float* __restrict__ in, __bf16* __restrict__ out, int n) {
    int i = blockIdx.x * blockDim.x + threadIdx.x;
    if (i < n) out[i] = f2bf(in[i]);
}

__global__ void zero_u32(unsigned* __restrict__ p, int n) {
    int i = blockIdx.x * blockDim.x + threadIdx.x;
    if (i < n) p[i] = 0u;
}

// ---------------- edge scatter-add: agg[dst] += x[src] ----------------
// One block per edge, one lane per feature -> coalesced loads, HW f32 fadd atomics.
template<int D>
__global__ void scatter_add_kernel(const float* __restrict__ x,
                                   const long long* __restrict__ src,
                                   const long long* __restrict__ dst,
                                   float* __restrict__ agg) {
    int e = blockIdx.x;
    int f = threadIdx.x;
    long long s = src[e];
    long long d = dst[e];
    unsafeAtomicAdd(&agg[d * (long long)D + f], x[s * (long long)D + f]);
}

// ---------------- weight pre-pack into WMMA B-fragment layout ----------------
// W is (K x M) row-major f32.  Packed layout: [kt][nt][lane][16 bf16], where for
// lane L: n = nt*16 + (L&15), k = kt*32 + ((L>>4)*16) + e   (e = 0..15)
// Contiguous k order, same as the A fragments (lanes 0-15: K 0..15, lanes 16-31: K 16..31).
__global__ void pack_weights(const float* __restrict__ W, __bf16* __restrict__ Wp, int K, int M) {
    int t = blockIdx.x * blockDim.x + threadIdx.x;
    int NT = M >> 4;
    int total = (K >> 5) * NT * 32;
    if (t >= total) return;
    int lane = t & 31;
    int nt = (t >> 5) % NT;
    int kt = (t >> 5) / NT;
    int n  = nt * 16 + (lane & 15);
    int ks = kt * 32 + ((lane >> 4) << 4);
    v16bf v;
#pragma unroll
    for (int e = 0; e < 16; ++e) v[e] = f2bf(W[(ks + e) * M + n]);
    ((v16bf*)Wp)[t] = v;
}

// ---------------- WMMA GEMM fragment helpers ----------------
// A is bf16 row-major; each lane loads two contiguous
// 8-bf16 (16B) chunks:  k = kt*32 + (lane>=16 ? 16 : 0) + {0..7, 8..15}.
__device__ __forceinline__ void load_a_frag(const __bf16* __restrict__ Arow, int kt, int half,
                                            v16bf& a) {
    const int kbase = kt * 32 + half * 16;
    v8bf lo = *(const v8bf*)(Arow + kbase);
    v8bf hi = *(const v8bf*)(Arow + kbase + 8);
#pragma unroll
    for (int e = 0; e < 8; ++e) { a[e] = lo[e]; a[e + 8] = hi[e]; }
}

__device__ __forceinline__ void load_b_frag(const v16bf* __restrict__ Wbase, int kt, int NT,
                                            v16bf (&b)[4]) {
    const v16bf* wp = Wbase + (long long)kt * NT * 32;
#pragma unroll
    for (int j = 0; j < 4; ++j) b[j] = wp[j * 32];
}

__device__ __forceinline__ void wmma_group(v8f (&acc)[4], const v16bf& a, const v16bf (&b)[4]) {
#pragma unroll
    for (int j = 0; j < 4; ++j) {
        acc[j] = __builtin_amdgcn_wmma_f32_16x16x32_bf16(
            false, a, false, b[j], (short)0, acc[j], false, false);
    }
}

// ---------------- WMMA GEMM:  out = relu(A[nrows x K] @ Wpacked + bias) ----------------
// Block = 128 threads = 4 waves; wave owns a 16(M) x 64(N) tile => 4 accumulators.
// Two-stage software pipeline with manual x2 unroll (KT is always even): loads write
// directly into the stage registers so there are no loop-carried copies, and each
// WMMA group consumes fragments loaded a full stage earlier (10 newer loads in flight).
__global__ void __launch_bounds__(128)
wmma_gemm_bias_relu(const __bf16* __restrict__ A, int K,
                    const __bf16* __restrict__ Wp,
                    const float* __restrict__ bias,
                    __bf16* __restrict__ outBf,
                    float*  __restrict__ outF32,
                    int nrows, int ncols) {
    const int lane = threadIdx.x & 31;
    const int wave = threadIdx.x >> 5;
    const int m0   = blockIdx.x * 64 + wave * 16;
    if (m0 >= nrows) return;                 // uniform per wave: EXEC stays all-ones for WMMA
    const int n0   = blockIdx.y * 64;
    const int half = lane >> 4;
    const int mrow = m0 + (lane & 15);
    const int NT   = ncols >> 4;
    const int KT   = K >> 5;                 // 4 or 8 (even)
    const int nt0  = n0 >> 4;

    const __bf16* Arow  = A + (long long)mrow * K;
    const v16bf*  Wbase = (const v16bf*)Wp + (long long)nt0 * 32 + lane;

    v8f acc[4] = {};

    v16bf a0, a1;
    v16bf b0[4], b1[4];

    load_a_frag(Arow, 0, half, a0);
    load_b_frag(Wbase, 0, NT, b0);

    for (int kt = 0; kt < KT; kt += 2) {
        // stage 1 loads (kt+1 < KT always, KT even)
        load_a_frag(Arow, kt + 1, half, a1);
        load_b_frag(Wbase, kt + 1, NT, b1);

        wmma_group(acc, a0, b0);

        // stage 0 loads for kt+2 (clamped on last iteration; harmless reload)
        const int k2 = (kt + 2 < KT) ? (kt + 2) : (KT - 1);
        load_a_frag(Arow, k2, half, a0);
        load_b_frag(Wbase, k2, NT, b0);

        wmma_group(acc, a1, b1);
    }

    // Epilogue: C layout -> VGPR r holds row m0 + r + half*8, column n0 + j*16 + (lane&15)
    const int col   = lane & 15;
    const int rbase = m0 + half * 8;
#pragma unroll
    for (int j = 0; j < 4; ++j) {
        const int n = n0 + j * 16 + col;
        const float bv = bias[n];
#pragma unroll
        for (int r = 0; r < 8; ++r) {
            float v = acc[j][r] + bv;
            v = v > 0.0f ? v : 0.0f;
            long long idx = (long long)(rbase + r) * ncols + n;
            if (outF32) outF32[idx] = v;
            if (outBf)  outBf[idx]  = f2bf(v);
        }
    }
}

// ---------------- global mean pool ----------------
__global__ void pool_scatter(const float* __restrict__ h, const long long* __restrict__ batch,
                             float* __restrict__ sums, int* __restrict__ counts) {
    int node = blockIdx.x;
    int f    = threadIdx.x;           // blockDim = HID
    long long g = batch[node];
    unsafeAtomicAdd(&sums[g * HID + f], h[(long long)node * HID + f]);
    if (f == 0) atomicAdd(&counts[g], 1);
}

__global__ void pool_final(const float* __restrict__ sums, const int* __restrict__ counts,
                           float* __restrict__ out) {
    int i = blockIdx.x * blockDim.x + threadIdx.x;
    if (i >= N_GRAPHS * HID) return;
    int g = i / HID;
    float c = (float)counts[g];
    out[i] = sums[i] / fmaxf(c, 1.0f);
}

// ---------------- workspace layout (bytes) ----------------
static constexpr size_t SZ_F32_NH = (size_t)N_NODES * HID * 4;   // 51.2 MB
static constexpr size_t SZ_BF_NH  = (size_t)N_NODES * HID * 2;   // 25.6 MB
static constexpr size_t OFF_A = 0;                               // f32 N*H : agg0 -> agg1 -> h2
static constexpr size_t OFF_B = OFF_A + SZ_F32_NH;               // f32 N*H : h0 (layer0 MLP out)
static constexpr size_t OFF_C = OFF_B + SZ_F32_NH;               // bf16 N*H: Abf0 -> Abf1
static constexpr size_t OFF_D = OFF_C + SZ_BF_NH;                // bf16 N*H: t1 -> t2
static constexpr size_t OFF_W = OFF_D + SZ_BF_NH;                // packed weights
static constexpr size_t SZ_W10 = (size_t)D_IN * HID * 2;
static constexpr size_t SZ_WHH = (size_t)HID * HID * 2;
static constexpr size_t OFF_P = OFF_W + SZ_W10 + 3 * SZ_WHH;     // pool sums + counts

extern "C" void kernel_launch(void* const* d_in, const int* in_sizes, int n_in,
                              void* d_out, int out_size, void* d_ws, size_t ws_size,
                              hipStream_t stream) {
    const float*     x     = (const float*)d_in[0];
    const long long* ei    = (const long long*)d_in[1];
    const long long* batch = (const long long*)d_in[2];
    const float* W1_0 = (const float*)d_in[3];
    const float* b1_0 = (const float*)d_in[4];
    const float* W2_0 = (const float*)d_in[5];
    const float* b2_0 = (const float*)d_in[6];
    const float* W1_1 = (const float*)d_in[7];
    const float* b1_1 = (const float*)d_in[8];
    const float* W2_1 = (const float*)d_in[9];
    const float* b2_1 = (const float*)d_in[10];

    char* ws = (char*)d_ws;
    float*  bufA = (float*)(ws + OFF_A);
    float*  bufB = (float*)(ws + OFF_B);
    __bf16* bufC = (__bf16*)(ws + OFF_C);
    __bf16* bufD = (__bf16*)(ws + OFF_D);
    __bf16* Wp10 = (__bf16*)(ws + OFF_W);
    __bf16* Wp20 = Wp10 + (size_t)D_IN * HID;
    __bf16* Wp11 = Wp20 + (size_t)HID * HID;
    __bf16* Wp21 = Wp11 + (size_t)HID * HID;
    float*  sums   = (float*)(ws + OFF_P);
    int*    counts = (int*)(ws + OFF_P + (size_t)N_GRAPHS * HID * 4);

    const long long* src = ei;
    const long long* dst = ei + N_EDGES;

    // --- pack weights into WMMA B-fragment layout (tiny, reused 50000x) ---
    pack_weights<<<8,  256, 0, stream>>>(W1_0, Wp10, D_IN, HID);   // (128/32)*(256/16)*32 = 2048
    pack_weights<<<16, 256, 0, stream>>>(W2_0, Wp20, HID,  HID);   // 4096
    pack_weights<<<16, 256, 0, stream>>>(W1_1, Wp11, HID,  HID);
    pack_weights<<<16, 256, 0, stream>>>(W2_1, Wp21, HID,  HID);

    // --- layer 0: agg0 = x; agg0[dst] += x[src] ---
    copy_f32_vec4<<<(N_NODES * D_IN / 4 + 255) / 256, 256, 0, stream>>>(x, bufA, N_NODES * D_IN / 4);
    scatter_add_kernel<D_IN><<<N_EDGES, D_IN, 0, stream>>>(x, src, dst, bufA);
    cvt_f32_bf16<<<(N_NODES * D_IN + 255) / 256, 256, 0, stream>>>(bufA, bufC, N_NODES * D_IN);

    dim3 ggrid((N_NODES + 63) / 64, HID / 64);
    // h = relu(agg @ W1_0 + b1_0)  -> bf16
    wmma_gemm_bias_relu<<<ggrid, 128, 0, stream>>>(bufC, D_IN, Wp10, b1_0, bufD, nullptr, N_NODES, HID);
    // h0 = relu(h @ W2_0 + b2_0)   -> f32 (needed for layer-1 aggregation)
    wmma_gemm_bias_relu<<<ggrid, 128, 0, stream>>>(bufD, HID,  Wp20, b2_0, nullptr, bufB, N_NODES, HID);

    // --- layer 1: agg1 = h0; agg1[dst] += h0[src] ---
    copy_f32_vec4<<<(N_NODES * HID / 4 + 255) / 256, 256, 0, stream>>>(bufB, bufA, N_NODES * HID / 4);
    scatter_add_kernel<HID><<<N_EDGES, HID, 0, stream>>>(bufB, src, dst, bufA);
    cvt_f32_bf16<<<(N_NODES * HID + 255) / 256, 256, 0, stream>>>(bufA, bufC, N_NODES * HID);

    wmma_gemm_bias_relu<<<ggrid, 128, 0, stream>>>(bufC, HID, Wp11, b1_1, bufD, nullptr, N_NODES, HID);
    // h2 -> f32 in bufA (agg1 is dead now)
    wmma_gemm_bias_relu<<<ggrid, 128, 0, stream>>>(bufD, HID, Wp21, b2_1, nullptr, bufA, N_NODES, HID);

    // --- global mean pool ---
    int poolWords = N_GRAPHS * HID + N_GRAPHS;   // sums + counts
    zero_u32<<<(poolWords + 255) / 256, 256, 0, stream>>>((unsigned*)sums, poolWords);
    pool_scatter<<<N_NODES, HID, 0, stream>>>(bufA, batch, sums, counts);
    pool_final<<<(N_GRAPHS * HID + 255) / 256, 256, 0, stream>>>(sums, counts, (float*)d_out);
}
